// WindowMultiHeadSelfAttention_60198261620816
// MI455X (gfx1250) — matrix operand order repaired and checked by reading, not compiled, not run on hardware
//
#include <hip/hip_runtime.h>

typedef __attribute__((ext_vector_type(16))) _Float16 v16h;
typedef __attribute__((ext_vector_type(8)))  _Float16 v8h;
typedef __attribute__((ext_vector_type(8)))  float    v8f;

#define N_HEADS 8
#define HEAD_DIM 32
#define DIM 256
#define LSEQ 49
#define LPAD 64
#define QKVN 768

union Frag  { v16h v; _Float16 h[16]; v8h q[2]; };
union FragF { v8f  v; float f[8]; };

__device__ __forceinline__ v8f wmma_f16(v16h a, v16h b, v8f c) {
  return __builtin_amdgcn_wmma_f32_16x16x32_f16(false, a, false, b, (short)0, c, false, false);
}

// A-fragment (16x32 f16, M x K) from row-major storage (stride in halves).
// lane<16: K {0..7,16..23}; lane>=16: K {8..15,24..31}; row = row0 + (lane&15)
__device__ __forceinline__ v16h load_a16(const _Float16* base, int row0, int kbase,
                                         int stride, int lane) {
  const _Float16* p = base + (row0 + (lane & 15)) * stride + kbase + ((lane >> 4) << 3);
  Frag f;
  f.q[0] = *(const v8h*)p;
  f.q[1] = *(const v8h*)(p + 16);
  return f.v;
}

// B-fragment (32x16 f16, K x N) where column n is stored contiguously
// (base[n*stride + k]). lane<16: K {0..7,16..23}; lane>=16: K {8..15,24..31}; n = n0 + (lane&15)
__device__ __forceinline__ v16h load_b16_col(const _Float16* base, int n0, int kbase,
                                             int stride, int lane) {
  const _Float16* p = base + (n0 + (lane & 15)) * stride + kbase + ((lane >> 4) << 3);
  Frag f;
  f.q[0] = *(const v8h*)p;
  f.q[1] = *(const v8h*)(p + 16);
  return f.v;
}

__global__ void convert_weights_kernel(const float* __restrict__ qkv_w,
                                       const float* __restrict__ proj_w,
                                       _Float16* __restrict__ wqkvT,
                                       _Float16* __restrict__ wprojT) {
  int idx = blockIdx.x * blockDim.x + threadIdx.x;
  const int NQ = QKVN * DIM;
  if (idx < NQ) {
    int n = idx / DIM, k = idx % DIM;
    wqkvT[idx] = (_Float16)qkv_w[k * QKVN + n];        // wqkvT[n][k]
  } else if (idx < NQ + DIM * DIM) {
    int i = idx - NQ;
    int n = i / DIM, k = i % DIM;
    wprojT[i] = (_Float16)proj_w[k * DIM + n];         // wprojT[n][k]
  }
}

__launch_bounds__(256)
__global__ void win_mhsa_kernel(const float* __restrict__ x,
                                const float* __restrict__ qkv_b,
                                const float* __restrict__ proj_b,
                                const _Float16* __restrict__ wqkvT,
                                const _Float16* __restrict__ wprojT,
                                float* __restrict__ out) {
  __shared__ _Float16 q_s[LPAD * DIM];    // 32 KB: Q [token][d]; reused as proj input in phase 5
  __shared__ _Float16 k_s[LPAD * DIM];    // 32 KB: K [token][d]
  __shared__ _Float16 vT_s[DIM * LPAD];   // 32 KB: V transposed [d][token]
  __shared__ _Float16 scr_s[N_HEADS * LPAD * LPAD]; // 64 KB: x tile, then attn per head

  const int b    = blockIdx.x;
  const int tid  = threadIdx.x;
  const int lane = tid & 31;
  const int wave = tid >> 5;
  const int hi8  = (lane >> 4) << 3;   // 0 or 8: row offset in C/D fragment
  const int cn   = lane & 15;          // column within 16-wide tile

  _Float16* xs = scr_s;                // [64][256] f16, rows 49..63 zero

  // ---- Phase 0: stage x window into LDS as f16 ----
  const float* xb = x + (size_t)b * LSEQ * DIM;
  for (int i = tid; i < LPAD * (DIM / 4); i += 256) {
    int row = i >> 6;
    int c4  = (i & 63) << 2;
    float4 v = make_float4(0.f, 0.f, 0.f, 0.f);
    if (row < LSEQ) v = *(const float4*)(xb + row * DIM + c4);
    _Float16* d = xs + row * DIM + c4;
    d[0] = (_Float16)v.x; d[1] = (_Float16)v.y;
    d[2] = (_Float16)v.z; d[3] = (_Float16)v.w;
  }
  __syncthreads();

  // ---- Phase 1: QKV = Xs @ Wqkv + b  (M=64, N=768, K=256) ----
  // Q tiles -> q_s, K tiles -> k_s, V tiles -> vT_s (transposed on store).
  for (int nt = wave * 6; nt < wave * 6 + 6; ++nt) {
    int n0 = nt << 4;
    float bias = qkv_b[n0 + cn];
    for (int mt = 0; mt < 4; ++mt) {
      // Batch all A loads (one LDS clause, one wait), then issue WMMAs back-to-back.
      v16h a[8];
#pragma unroll
      for (int kk = 0; kk < 8; ++kk)
        a[kk] = load_a16(xs, mt << 4, kk << 5, DIM, lane);
      v8f acc = {};
#pragma unroll
      for (int kk = 0; kk < 8; ++kk) {
        v16h bm = load_b16_col(wqkvT, n0, kk << 5, DIM, lane);
        acc = wmma_f16(a[kk], bm, acc);
      }
      FragF af; af.v = acc;
      int row0 = (mt << 4) + hi8;
      if (n0 < DIM) {                      // Q
        int col = n0 + cn;
#pragma unroll
        for (int r = 0; r < 8; ++r)
          q_s[(row0 + r) * DIM + col] = (_Float16)(af.f[r] + bias);
      } else if (n0 < 2 * DIM) {           // K
        int col = n0 - DIM + cn;
#pragma unroll
        for (int r = 0; r < 8; ++r)
          k_s[(row0 + r) * DIM + col] = (_Float16)(af.f[r] + bias);
      } else {                             // V, transposed: 8 consecutive halves
        int drow = n0 - 2 * DIM + cn;
        _Float16* vp = vT_s + drow * LPAD + row0;
#pragma unroll
        for (int r = 0; r < 8; ++r)
          vp[r] = (_Float16)(af.f[r] + bias);
      }
    }
  }
  __syncthreads();

  // ---- Phase 2: per-head scores + softmax (wave == head) ----
  const _Float16* qh  = q_s + wave * HEAD_DIM;
  const _Float16* kh  = k_s + wave * HEAD_DIM;
  const _Float16* vhT = vT_s + wave * HEAD_DIM * LPAD;  // [32][64]

  // Preload all Q (A) and K^T (B) fragments, then 16 WMMAs back-to-back.
  v16h aq[4], bk[4];
#pragma unroll
  for (int mt = 0; mt < 4; ++mt) aq[mt] = load_a16(qh, mt << 4, 0, DIM, lane);
#pragma unroll
  for (int nt = 0; nt < 4; ++nt) bk[nt] = load_b16_col(kh, nt << 4, 0, DIM, lane);

  FragF sf[4][4];
#pragma unroll
  for (int nt = 0; nt < 4; ++nt)
#pragma unroll
    for (int mt = 0; mt < 4; ++mt)
      sf[mt][nt].v = wmma_f16(aq[mt], bk[nt], v8f{});

  const float scale = 0.17677669529663687f;  // 32^-0.5
#pragma unroll
  for (int mt = 0; mt < 4; ++mt) {
#pragma unroll
    for (int r = 0; r < 8; ++r) {
      float vals[4];
#pragma unroll
      for (int nt = 0; nt < 4; ++nt) {
        int col = (nt << 4) + cn;
        float v = sf[mt][nt].f[r] * scale;
        vals[nt] = (col < LSEQ) ? v : -1e30f;
      }
      float m = fmaxf(fmaxf(vals[0], vals[1]), fmaxf(vals[2], vals[3]));
#pragma unroll
      for (int sh = 1; sh < 16; sh <<= 1) m = fmaxf(m, __shfl_xor(m, sh, 32));
      float sum = 0.f;
#pragma unroll
      for (int nt = 0; nt < 4; ++nt) {
        float p = __expf(vals[nt] - m);
        vals[nt] = p; sum += p;
      }
#pragma unroll
      for (int sh = 1; sh < 16; sh <<= 1) sum += __shfl_xor(sum, sh, 32);
      float inv = 1.0f / sum;
#pragma unroll
      for (int nt = 0; nt < 4; ++nt) sf[mt][nt].f[r] = vals[nt] * inv;
    }
  }

  // Store attn as f16 into per-head scratch (x tile is dead now).
  _Float16* attn = scr_s + wave * (LPAD * LPAD);
#pragma unroll
  for (int mt = 0; mt < 4; ++mt)
#pragma unroll
    for (int nt = 0; nt < 4; ++nt)
#pragma unroll
      for (int r = 0; r < 8; ++r)
        attn[((mt << 4) + hi8 + r) * LPAD + (nt << 4) + cn] = (_Float16)sf[mt][nt].f[r];
  __syncthreads();   // after this: attn visible; all Q reads from q_s complete

  // ---- Phase 4: O = attn @ V  (M=64, N=32, K=64) ----
  // Preload all fragments (8 A + 4 B), then 16 WMMAs.
  v16h aa[2][4], bv[2][2];
#pragma unroll
  for (int kk = 0; kk < 2; ++kk) {
#pragma unroll
    for (int mt = 0; mt < 4; ++mt)
      aa[kk][mt] = load_a16(attn, mt << 4, kk << 5, LPAD, lane);
#pragma unroll
    for (int nt = 0; nt < 2; ++nt)
      bv[kk][nt] = load_b16_col(vhT, nt << 4, kk << 5, LPAD, lane);
  }

  v8f O[4][2];
#pragma unroll
  for (int mt = 0; mt < 4; ++mt)
#pragma unroll
    for (int nt = 0; nt < 2; ++nt) O[mt][nt] = v8f{};
#pragma unroll
  for (int kk = 0; kk < 2; ++kk)
#pragma unroll
    for (int nt = 0; nt < 2; ++nt)
#pragma unroll
      for (int mt = 0; mt < 4; ++mt)
        O[mt][nt] = wmma_f16(aa[kk][mt], bv[kk][nt], O[mt][nt]);

  // Store per-head output into q_s reused as [64][256] f16 (q_s dead since phase 2,
  // and the attn barrier above guarantees every wave is past its Q reads).
  _Float16* outs = q_s;
#pragma unroll
  for (int mt = 0; mt < 4; ++mt)
#pragma unroll
    for (int nt = 0; nt < 2; ++nt) {
      FragF of; of.v = O[mt][nt];
      int col = wave * HEAD_DIM + (nt << 4) + cn;
#pragma unroll
      for (int r = 0; r < 8; ++r)
        outs[((mt << 4) + hi8 + r) * DIM + col] = (_Float16)of.f[r];
    }
  __syncthreads();

  // ---- Phase 5: out = O @ Wproj + b  (M=64, N=256, K=256) ----
  float* ob = out + (size_t)b * LSEQ * DIM;
  for (int nt = wave * 2; nt < wave * 2 + 2; ++nt) {
    int n0 = nt << 4;
    float bias = proj_b[n0 + cn];
    for (int mt = 0; mt < 4; ++mt) {
      v16h a[8];
#pragma unroll
      for (int kk = 0; kk < 8; ++kk)
        a[kk] = load_a16(outs, mt << 4, kk << 5, DIM, lane);
      v8f acc = {};
#pragma unroll
      for (int kk = 0; kk < 8; ++kk) {
        v16h bw = load_b16_col(wprojT, n0, kk << 5, DIM, lane);
        acc = wmma_f16(a[kk], bw, acc);
      }
      int col  = n0 + cn;
      int row0 = (mt << 4) + hi8;
      FragF af; af.v = acc;
#pragma unroll
      for (int r = 0; r < 8; ++r) {
        int row = row0 + r;
        if (row < LSEQ) ob[row * DIM + col] = af.f[r] + bias;
      }
    }
  }
}

extern "C" void kernel_launch(void* const* d_in, const int* in_sizes, int n_in,
                              void* d_out, int out_size, void* d_ws, size_t ws_size,
                              hipStream_t stream) {
  const float* x      = (const float*)d_in[0];
  const float* qkv_w  = (const float*)d_in[1];
  const float* qkv_b  = (const float*)d_in[2];
  const float* proj_w = (const float*)d_in[3];
  const float* proj_b = (const float*)d_in[4];
  float* out = (float*)d_out;

  _Float16* wqkvT  = (_Float16*)d_ws;
  _Float16* wprojT = (_Float16*)((char*)d_ws + (size_t)QKVN * DIM * sizeof(_Float16));

  int B = in_sizes[0] / (LSEQ * DIM);   // 2048

  int total = QKVN * DIM + DIM * DIM;
  convert_weights_kernel<<<(total + 255) / 256, 256, 0, stream>>>(qkv_w, proj_w, wqkvT, wprojT);
  win_mhsa_kernel<<<B, 256, 0, stream>>>(x, qkv_b, proj_b, wqkvT, wprojT, out);
}
